// ChainOfExperts_76141180223614
// MI455X (gfx1250) — compile-verified, matrix operand order checked by reading, not run on hardware
//
#include <hip/hip_runtime.h>
#include <hip/hip_bf16.h>
#include <cstdint>
#include <cstddef>

// ---------------------------------------------------------------------------
// MoE (chain-of-experts) for MI455X / gfx1250, wave32, bf16 WMMA.
// Bandwidth-bound (~0.6 GB traffic @ 23.3 TB/s ~ 26us floor); bf16
// v_wmma_f32_16x16x32_bf16 makes the ~410 GFLOP of GEMM negligible.
// Sparse routing (top-2 of 16) avoids the reference's dense 8x flop waste.
// A tiles: CDNA5 async global->LDS DMA (ASYNCcnt), double buffered so global
// latency hides under the previous tile's WMMAs.  B tiles: 16B-coalesced
// global loads + transposed LDS stores (full-width load wavefronts).
// ---------------------------------------------------------------------------

typedef __attribute__((ext_vector_type(16))) __bf16 v16bf;
typedef __attribute__((ext_vector_type(8)))  __bf16 v8bf;
typedef __attribute__((ext_vector_type(8)))  float  v8f;

#define T_TOK 16384   // B*S
#define DIM   2048    // D
#define NEXP  16      // routed experts
#define NSHE  2       // shared experts
#define DRI   512     // routed inner dim
#define DSI   1024    // shared inner dim

#define BM 128
#define BN 128
#define BK 32
#define LDSTR (BK + 8)          // 40 elems = 80 bytes (16B aligned rows)
#define THREADS 256             // 8 waves of 32

// ---------------------------------------------------------------------------
// Utility kernels
// ---------------------------------------------------------------------------
__global__ void cast_f32_bf16(const float* __restrict__ s, __bf16* __restrict__ d, long n) {
  long i = (long)blockIdx.x * blockDim.x + threadIdx.x;
  long stride = (long)gridDim.x * blockDim.x;
  for (; i < n; i += stride) d[i] = (__bf16)s[i];
}

// out[t,d] = sh_b2[0][d] + sh_b2[1][d]   (bias of shared experts, summed)
__global__ void init_out(float* __restrict__ out, const float* __restrict__ sh_b2, long n) {
  long i = (long)blockIdx.x * blockDim.x + threadIdx.x;
  long stride = (long)gridDim.x * blockDim.x;
  for (; i < n; i += stride) {
    int d = (int)(i % DIM);
    out[i] = sh_b2[d] + sh_b2[DIM + d];
  }
}

__global__ void zero_counts(int* __restrict__ counts) {
  if (threadIdx.x < NEXP) counts[threadIdx.x] = 0;
}

// logits[t,e] = dot(x[t,:], router_w[step, :, e])
__global__ void router_logits(const float* __restrict__ x, const float* __restrict__ rw,
                              const int* __restrict__ step_ptr, float* __restrict__ logits) {
  int i = blockIdx.x * blockDim.x + threadIdx.x;
  if (i >= T_TOK * NEXP) return;
  int t = i / NEXP, e = i % NEXP;
  const float* rwe = rw + (size_t)step_ptr[0] * DIM * NEXP;
  const float* xt  = x + (size_t)t * DIM;
  float s = 0.f;
  for (int d = 0; d < DIM; ++d) s += xt[d] * rwe[(size_t)d * NEXP + e];
  logits[i] = s;
}

// softmax over 16 logits, pick top-2 probs, append (token, prob) to expert list
__global__ void router_topk(const float* __restrict__ logits, int* __restrict__ counts,
                            int* __restrict__ tok_idx, float* __restrict__ tok_w) {
  int t = blockIdx.x * blockDim.x + threadIdx.x;
  if (t >= T_TOK) return;
  float p[NEXP];
  float m = -1e30f;
  for (int e = 0; e < NEXP; ++e) { p[e] = logits[(size_t)t * NEXP + e]; m = p[e] > m ? p[e] : m; }
  float s = 0.f;
  for (int e = 0; e < NEXP; ++e) { p[e] = __expf(p[e] - m); s += p[e]; }
  float inv = 1.f / s;
  int i1 = 0; float p1 = -1.f;
  for (int e = 0; e < NEXP; ++e) { float pe = p[e] * inv; if (pe > p1) { p1 = pe; i1 = e; } }
  int i2 = 0; float p2 = -1.f;
  for (int e = 0; e < NEXP; ++e) {
    if (e == i1) continue;
    float pe = p[e] * inv; if (pe > p2) { p2 = pe; i2 = e; }
  }
  int pos1 = atomicAdd(&counts[i1], 1);
  tok_idx[(size_t)i1 * T_TOK + pos1] = t;
  tok_w  [(size_t)i1 * T_TOK + pos1] = p1;
  int pos2 = atomicAdd(&counts[i2], 1);
  tok_idx[(size_t)i2 * T_TOK + pos2] = t;
  tok_w  [(size_t)i2 * T_TOK + pos2] = p2;
}

// low 32 bits of a generic pointer to __shared__ == wave-relative LDS offset
// (ISA flat aperture)
__device__ __forceinline__ uint32_t lds_offset(const void* p) {
  return (uint32_t)(uintptr_t)p;
}

// ---------------------------------------------------------------------------
// Tiled bf16 WMMA GEMM.  C(BMxBN) tile per block; 8 waves each own 32x64.
// UP:  H = silu(A_gathered @ W + b1)   -> bf16 (compact rows)
// DOWN: out[token] += scale * (A @ W + b2)  (fp32 scatter-accumulate)
// A tiles: async global->LDS (double buffered, issued one K-step ahead).
// B tiles: 16B global loads + transposed ds stores, also one step ahead.
// ---------------------------------------------------------------------------
template <bool UP>
__global__ __launch_bounds__(THREADS)
void moe_gemm(const __bf16* __restrict__ A, int lda, int K, int N,
              const __bf16* __restrict__ W,
              const float*  __restrict__ bias,
              __bf16* __restrict__ Hout,
              float*  __restrict__ Out,
              const int*   __restrict__ gather,
              const float* __restrict__ rowscale,
              const int*   __restrict__ countptr, int fixedcount) {
  int rowcount = countptr ? countptr[0] : fixedcount;
  int m0 = blockIdx.x * BM;
  if (m0 >= rowcount) return;          // block-uniform early exit
  int n0 = blockIdx.y * BN;

  __shared__ __align__(16) __bf16 As[2][BM * LDSTR];   // [row][k]
  __shared__ __align__(16) __bf16 Bs[2][BN * LDSTR];   // [col][k] (transposed)

  int tid  = threadIdx.x;
  int lane = tid & 31;
  int wid  = tid >> 5;
  int wm   = wid & 3;    // 4 wave rows  -> 32 output rows each
  int wn   = wid >> 2;   // 2 wave cols  -> 64 output cols each

  // ---- stage A tile via async global->LDS DMA (no VGPR round trip) ----
  auto stageA = [&](int k0, int buf) {
    for (int p = 0; p < 2; ++p) {
      int linear = tid + p * THREADS;        // 0..511
      int row    = linear >> 2;
      int chunk  = linear & 3;
      __bf16* dst = &As[buf][row * LDSTR + chunk * 8];
      int grow = m0 + row;
      if (grow < rowcount) {
        int g = (UP && gather) ? gather[grow] : grow;
        uint64_t ga = (uint64_t)(uintptr_t)(A + (size_t)g * lda + k0 + chunk * 8);
        uint32_t lo = lds_offset(dst);
        asm volatile("global_load_async_to_lds_b128 %0, %1, off"
                     :: "v"(lo), "v"(ga) : "memory");
      } else {
        uint4 z; z.x = z.y = z.z = z.w = 0u;
        *(uint4*)dst = z;
      }
    }
  };

  // ---- stage W tile transposed: 16B-coalesced loads, b16 transposed stores
  //      Bs[n][k] = W[k0+k][n0+n]; lane-consecutive 8-elem chunks along N ----
  auto stageB = [&](int k0, int buf) {
    for (int p = 0; p < 2; ++p) {
      int linear = tid + p * THREADS;        // 0..511 chunks of 8 elements
      int k  = linear >> 4;                  // 16 chunks per K-row
      int nc = (linear & 15) * 8;            // N offset of this chunk
      v8bf v = *(const v8bf*)(W + (size_t)(k0 + k) * N + (n0 + nc));
      #pragma unroll
      for (int j = 0; j < 8; ++j)
        Bs[buf][(nc + j) * LDSTR + k] = v[j];
    }
  };

  v8f acc[2][4];
  for (int mi = 0; mi < 2; ++mi)
    for (int ni = 0; ni < 4; ++ni)
      for (int r = 0; r < 8; ++r) acc[mi][ni][r] = 0.f;

  const int frow  = lane & 15;
  const int khalf = (lane >> 4) * 8;   // ISA 16-bit A/B fragment layout

  // prologue: fill buffer 0
  stageA(0, 0);
  stageB(0, 0);
  asm volatile("s_wait_asynccnt 0x0" ::: "memory");
  __syncthreads();

  int cur = 0;
  for (int k0 = 0; k0 < K; k0 += BK) {
    int  nxt  = cur ^ 1;
    bool more = (k0 + BK) < K;

    // issue next A tile's async DMA before computing this one
    if (more) stageA(k0 + BK, nxt);

    // ---- fragments: two aligned 16B LDS reads per v16bf ----
    v16bf afrag[2], bfrag[4];
    for (int mi = 0; mi < 2; ++mi) {
      const __bf16* base = &As[cur][(wm * 32 + mi * 16 + frow) * LDSTR];
      v8bf lo = *(const v8bf*)(base + khalf);
      v8bf hi = *(const v8bf*)(base + 16 + khalf);
      afrag[mi] = __builtin_shufflevector(lo, hi, 0,1,2,3,4,5,6,7,8,9,10,11,12,13,14,15);
    }
    for (int ni = 0; ni < 4; ++ni) {
      const __bf16* base = &Bs[cur][(wn * 64 + ni * 16 + frow) * LDSTR];
      v8bf lo = *(const v8bf*)(base + khalf);
      v8bf hi = *(const v8bf*)(base + 16 + khalf);
      bfrag[ni] = __builtin_shufflevector(lo, hi, 0,1,2,3,4,5,6,7,8,9,10,11,12,13,14,15);
    }

    for (int mi = 0; mi < 2; ++mi)
      for (int ni = 0; ni < 4; ++ni)
        acc[mi][ni] = __builtin_amdgcn_wmma_f32_16x16x32_bf16(
            false, afrag[mi], false, bfrag[ni], (short)0, acc[mi][ni], false, false);

    // overlap B staging with the WMMAs above, then drain async + barrier
    if (more) {
      stageB(k0 + BK, nxt);
      asm volatile("s_wait_asynccnt 0x0" ::: "memory");
    }
    __syncthreads();
    cur = nxt;
  }

  // ---- epilogue (C/D layout: vgpr r, lane<16 -> M=r, lane>=16 -> M=r+8) ----
  int ncol_l = lane & 15;
  int moff   = (lane >> 4) * 8;
  for (int mi = 0; mi < 2; ++mi) {
    for (int ni = 0; ni < 4; ++ni) {
      int col = n0 + wn * 64 + ni * 16 + ncol_l;
      float b = bias ? bias[col] : 0.f;
      for (int r = 0; r < 8; ++r) {
        int row = m0 + wm * 32 + mi * 16 + moff + r;
        if (UP) {
          float v = acc[mi][ni][r] + b;
          v = v * (1.f / (1.f + __expf(-v)));     // silu
          Hout[(size_t)row * N + col] = (__bf16)v;
        } else {
          if (row < rowcount) {
            int   t = gather ? gather[row] : row;
            float s = rowscale ? rowscale[row] : 1.f;
            Out[(size_t)t * N + col] += s * (acc[mi][ni][r] + b);
          }
        }
      }
    }
  }
}

// ---------------------------------------------------------------------------
// Host launcher
// ---------------------------------------------------------------------------
extern "C" void kernel_launch(void* const* d_in, const int* in_sizes, int n_in,
                              void* d_out, int out_size, void* d_ws, size_t ws_size,
                              hipStream_t stream) {
  (void)in_sizes; (void)n_in; (void)out_size; (void)ws_size;

  const float* x        = (const float*)d_in[0];
  const int*   step_t   = (const int*)  d_in[1];
  const float* router_w = (const float*)d_in[2];
  const float* re_w1    = (const float*)d_in[3];
  const float* re_b1    = (const float*)d_in[4];
  const float* re_w2    = (const float*)d_in[5];
  const float* re_b2    = (const float*)d_in[6];
  const float* sh_w1    = (const float*)d_in[7];
  const float* sh_b1    = (const float*)d_in[8];
  const float* sh_w2    = (const float*)d_in[9];
  const float* sh_b2    = (const float*)d_in[10];
  float* out = (float*)d_out;

  // workspace carve-out (~188 MB)
  uint8_t* ws = (uint8_t*)d_ws;
  size_t off = 0;
  auto carve = [&](size_t bytes) -> void* {
    void* p = ws + off;
    off += (bytes + 255) & ~(size_t)255;
    return p;
  };
  __bf16* xbf    = (__bf16*)carve((size_t)T_TOK * DIM * 2);
  __bf16* shw1bf = (__bf16*)carve((size_t)NSHE * DIM * DSI * 2);
  __bf16* shw2bf = (__bf16*)carve((size_t)NSHE * DSI * DIM * 2);
  __bf16* rew1bf = (__bf16*)carve((size_t)NEXP * DIM * DRI * 2);
  __bf16* rew2bf = (__bf16*)carve((size_t)NEXP * DRI * DIM * 2);
  __bf16* Hbuf   = (__bf16*)carve((size_t)T_TOK * DSI * 2);      // reused per expert
  float*  logits = (float*) carve((size_t)T_TOK * NEXP * 4);
  int*    counts = (int*)   carve(256);
  int*    tok_idx= (int*)   carve((size_t)NEXP * T_TOK * 4);
  float*  tok_w  = (float*) carve((size_t)NEXP * T_TOK * 4);

  // 1) bf16 casts (once per call; amortized vs fp32 GEMM traffic)
  auto cast = [&](const float* s, __bf16* d, long n) {
    cast_f32_bf16<<<2048, 256, 0, stream>>>(s, d, n);
  };
  cast(x,     xbf,    (long)T_TOK * DIM);
  cast(sh_w1, shw1bf, (long)NSHE * DIM * DSI);
  cast(sh_w2, shw2bf, (long)NSHE * DSI * DIM);
  cast(re_w1, rew1bf, (long)NEXP * DIM * DRI);
  cast(re_w2, rew2bf, (long)NEXP * DRI * DIM);

  // 2) out = sum of shared down-proj biases
  init_out<<<4096, 256, 0, stream>>>(out, sh_b2, (long)T_TOK * DIM);

  // 3) routing
  zero_counts<<<1, 32, 0, stream>>>(counts);
  router_logits<<<(T_TOK * NEXP + 255) / 256, 256, 0, stream>>>(x, router_w, step_t, logits);
  router_topk<<<(T_TOK + 255) / 256, 256, 0, stream>>>(logits, counts, tok_idx, tok_w);

  // 4) shared experts (dense, all tokens), sequential so out += is race-free
  for (int s = 0; s < NSHE; ++s) {
    moe_gemm<true><<<dim3(T_TOK / BM, DSI / BN), THREADS, 0, stream>>>(
        xbf, DIM, DIM, DSI, shw1bf + (size_t)s * DIM * DSI, sh_b1 + (size_t)s * DSI,
        Hbuf, nullptr, nullptr, nullptr, nullptr, T_TOK);
    moe_gemm<false><<<dim3(T_TOK / BM, DIM / BN), THREADS, 0, stream>>>(
        Hbuf, DSI, DSI, DIM, shw2bf + (size_t)s * DSI * DIM, nullptr,
        nullptr, out, nullptr, nullptr, nullptr, T_TOK);
  }

  // 5) routed experts: gather -> GEMM -> silu -> GEMM -> weighted scatter.
  //    Per-expert lists hold unique tokens and experts run back-to-back on
  //    the stream, so the fp32 scatter-accumulate needs no atomics.
  for (int e = 0; e < NEXP; ++e) {
    moe_gemm<true><<<dim3(T_TOK / BM, DRI / BN), THREADS, 0, stream>>>(
        xbf, DIM, DIM, DRI, rew1bf + (size_t)e * DIM * DRI, re_b1 + (size_t)e * DRI,
        Hbuf, nullptr, tok_idx + (size_t)e * T_TOK, nullptr, counts + e, 0);
    moe_gemm<false><<<dim3(T_TOK / BM, DIM / BN), THREADS, 0, stream>>>(
        Hbuf, DRI, DRI, DIM, rew2bf + (size_t)e * DRI * DIM, re_b2 + (size_t)e * DIM,
        nullptr, out, tok_idx + (size_t)e * T_TOK, tok_w + (size_t)e * T_TOK, counts + e, 0);
  }
}
